// GCNEncoder_7413113553701
// MI455X (gfx1250) — matrix operand order repaired and checked by reading, not compiled, not run on hardware
//
#include <hip/hip_runtime.h>
#include <hip/hip_bf16.h>

typedef __attribute__((ext_vector_type(16))) _Float16 v16h;
typedef __attribute__((ext_vector_type(8)))  float    v8f;

#define IN_CH 128
#define HID   128
#define OUTC  64

// ---------------------------------------------------------------------------
// Degree / normalization kernels
// ---------------------------------------------------------------------------
__global__ void k_deg_init(float* __restrict__ deg, int n) {
  int i = blockIdx.x * blockDim.x + threadIdx.x;
  if (i < n) deg[i] = 1.0f;              // self loop contributes 1
}

__global__ void k_deg_accum(const long long* __restrict__ dst, float* __restrict__ deg, int ne) {
  int e = blockIdx.x * blockDim.x + threadIdx.x;
  if (e < ne) atomicAdd(&deg[(int)dst[e]], 1.0f);
}

__global__ void k_dinv(const float* __restrict__ deg, float* __restrict__ dinv, int n) {
  int i = blockIdx.x * blockDim.x + threadIdx.x;
  if (i < n) dinv[i] = rsqrtf(deg[i]);   // deg >= 1 always (self loop)
}

__global__ void k_coef(const long long* __restrict__ src, const long long* __restrict__ dst,
                       const float* __restrict__ dinv, float* __restrict__ coef, int ne) {
  int e = blockIdx.x * blockDim.x + threadIdx.x;
  if (e < ne) coef[e] = dinv[(int)src[e]] * dinv[(int)dst[e]];
}

// ---------------------------------------------------------------------------
// WMMA GEMM:  C[nrows x NCOL] = A[nrows x 128] * W[128 x NCOL]
// One wave (32 threads) per 16x16 output tile; K = 128 -> 4 x wmma 16x16x32.
// fp32 inputs are converted to f16 in-register while building the fragments.
//
// A fragment (16-bit A 16x32, ISA 7.12.2): lane = M + 16*(K-half-of-8),
//   VGPR j in 0..3 -> K = half*8 + 2j, VGPR j in 4..7 -> K = 16 + half*8 + 2(j-4)
// B fragment (16-bit B 32x16): lane = N + 16*(K-half-of-8),
//   VGPR j -> same K as the A fragment
// C/D (f32 16x16): lane = N + 16*sel, VGPR r -> M = r + 8*sel
// ---------------------------------------------------------------------------
template <int NCOL>
__global__ void k_gemm_wmma(const float* __restrict__ A, const float* __restrict__ W,
                            float* __restrict__ C, int nrows) {
  const int lane = threadIdx.x;           // 0..31
  const int tm   = blockIdx.x;            // m tile
  const int tn   = blockIdx.y;            // n tile
  const int l15  = lane & 15;
  const int half = lane >> 4;             // 0 or 1

  int rowA = tm * 16 + l15;
  if (rowA > nrows - 1) rowA = nrows - 1; // clamp reads; keep EXEC full for WMMA
  const int colB = tn * 16 + l15;

  v8f acc = {};
#pragma unroll
  for (int kk = 0; kk < IN_CH; kk += 32) {
    v16h a, b;
#pragma unroll
    for (int j = 0; j < 8; ++j) {
      const int ka = kk + ((j >> 2) << 4) + (half << 3) + ((j & 3) << 1);
      a[2 * j]     = (_Float16)A[rowA * IN_CH + ka];
      a[2 * j + 1] = (_Float16)A[rowA * IN_CH + ka + 1];
      const int kb = kk + ((j >> 2) << 4) + (half << 3) + ((j & 3) << 1);
      b[2 * j]     = (_Float16)W[kb * NCOL + colB];
      b[2 * j + 1] = (_Float16)W[(kb + 1) * NCOL + colB];
    }
    acc = __builtin_amdgcn_wmma_f32_16x16x32_f16(false, a, false, b, (short)0, acc,
                                                 false, false);
  }

#pragma unroll
  for (int r = 0; r < 8; ++r) {
    const int om = tm * 16 + r + (half << 3);
    if (om < nrows) C[(long long)om * NCOL + tn * 16 + l15] = acc[r];
  }
}

// ---------------------------------------------------------------------------
// out[i*F + c] = bias[c] + dinv[i]^2 * vw[i*F + c]   (bias + fused self-loop)
// ---------------------------------------------------------------------------
template <int F>
__global__ void k_bias_selfloop(const float* __restrict__ vw, const float* __restrict__ bias,
                                const float* __restrict__ dinv, float* __restrict__ out, int n) {
  const long long idx = (long long)blockIdx.x * blockDim.x + threadIdx.x;
  if (idx >= (long long)n * F) return;
  const int node = (int)(idx / F);
  const int c    = (int)(idx % F);
  const float di = dinv[node];
  out[idx] = bias[c] + di * di * vw[idx];
}

// ---------------------------------------------------------------------------
// Edge scatter: acc[dst, :] += coef[e] * val[src, :]
// 32 lanes per edge; F/32 floats per lane (vectorized load, scalar f32 atomics).
// ---------------------------------------------------------------------------
template <int F>
__global__ void k_scatter(const float* __restrict__ val, const long long* __restrict__ src,
                          const long long* __restrict__ dst, const float* __restrict__ coef,
                          float* __restrict__ acc, int ne) {
  const long long tid = (long long)blockIdx.x * blockDim.x + threadIdx.x;
  const int e    = (int)(tid >> 5);
  const int lane = (int)(tid & 31);
  if (e >= ne) return;
  const int   s  = (int)src[e];
  const int   d  = (int)dst[e];
  const float cf = coef[e];
  if (F == 128) {
    const float4 v = *(const float4*)(val + (long long)s * F + lane * 4);
    float* ap = acc + (long long)d * F + lane * 4;
    atomicAdd(ap + 0, v.x * cf);
    atomicAdd(ap + 1, v.y * cf);
    atomicAdd(ap + 2, v.z * cf);
    atomicAdd(ap + 3, v.w * cf);
  } else {
    const float2 v = *(const float2*)(val + (long long)s * F + lane * 2);
    float* ap = acc + (long long)d * F + lane * 2;
    atomicAdd(ap + 0, v.x * cf);
    atomicAdd(ap + 1, v.y * cf);
  }
}

// ---------------------------------------------------------------------------
// ReLU + row L2 normalize (F = 128): one wave32 per node, 4 floats per lane.
// ---------------------------------------------------------------------------
__global__ void k_relu_norm(float* __restrict__ h, int n) {
  const long long tid = (long long)blockIdx.x * blockDim.x + threadIdx.x;
  const int node = (int)(tid >> 5);
  const int lane = (int)(tid & 31);
  if (node >= n) return;
  float4* p = (float4*)(h + (long long)node * HID) + lane;
  float4 v = *p;
  v.x = fmaxf(v.x, 0.0f);
  v.y = fmaxf(v.y, 0.0f);
  v.z = fmaxf(v.z, 0.0f);
  v.w = fmaxf(v.w, 0.0f);
  float ss = v.x * v.x + v.y * v.y + v.z * v.z + v.w * v.w;
#pragma unroll
  for (int m = 16; m >= 1; m >>= 1) ss += __shfl_xor(ss, m, 32);
  const float inv = 1.0f / fmaxf(sqrtf(ss), 1e-12f);
  v.x *= inv; v.y *= inv; v.z *= inv; v.w *= inv;
  *p = v;
}

// ---------------------------------------------------------------------------
// Host-side orchestration
// ---------------------------------------------------------------------------
extern "C" void kernel_launch(void* const* d_in, const int* in_sizes, int n_in,
                              void* d_out, int out_size, void* d_ws, size_t ws_size,
                              hipStream_t stream) {
  const float*     x   = (const float*)d_in[0];
  const long long* ei  = (const long long*)d_in[1];
  const float*     W1  = (const float*)d_in[2];
  const float*     b1  = (const float*)d_in[3];
  const float*     W2  = (const float*)d_in[4];
  const float*     b2  = (const float*)d_in[5];
  float*           out = (float*)d_out;

  const int N = in_sizes[0] / IN_CH;
  const int E = in_sizes[1] / 2;
  const long long* src = ei;
  const long long* dst = ei + E;

  // Workspace carve-out (floats)
  float* ws   = (float*)d_ws;
  float* deg  = ws;                          // N
  float* dinv = deg + N;                     // N
  float* coef = dinv + N;                    // E
  float* xw   = coef + E;                    // N * HID
  float* h    = xw + (size_t)N * HID;        // N * HID  (accumulator, then normalized)
  float* hw   = h + (size_t)N * HID;         // N * OUTC

  const int B = 256;

  // --- normalization coefficients ---
  k_deg_init<<<(N + B - 1) / B, B, 0, stream>>>(deg, N);
  k_deg_accum<<<(E + B - 1) / B, B, 0, stream>>>(dst, deg, E);
  k_dinv<<<(N + B - 1) / B, B, 0, stream>>>(deg, dinv, N);
  k_coef<<<(E + B - 1) / B, B, 0, stream>>>(src, dst, dinv, coef, E);

  // --- layer 1: xw = X @ W1 (WMMA), agg, relu, L2-normalize ---
  {
    dim3 grid((N + 15) / 16, HID / 16);
    k_gemm_wmma<HID><<<grid, 32, 0, stream>>>(x, W1, xw, N);
  }
  {
    long long tot = (long long)N * HID;
    k_bias_selfloop<HID><<<(int)((tot + B - 1) / B), B, 0, stream>>>(xw, b1, dinv, h, N);
  }
  {
    long long tot = (long long)E * 32;
    k_scatter<HID><<<(int)((tot + B - 1) / B), B, 0, stream>>>(xw, src, dst, coef, h, E);
  }
  {
    long long tot = (long long)N * 32;
    k_relu_norm<<<(int)((tot + B - 1) / B), B, 0, stream>>>(h, N);
  }

  // --- layer 2: hw = h @ W2 (WMMA), agg into d_out ---
  {
    dim3 grid((N + 15) / 16, OUTC / 16);
    k_gemm_wmma<OUTC><<<grid, 32, 0, stream>>>(h, W2, hw, N);
  }
  {
    long long tot = (long long)N * OUTC;
    k_bias_selfloop<OUTC><<<(int)((tot + B - 1) / B), B, 0, stream>>>(hw, b2, dinv, out, N);
  }
  {
    long long tot = (long long)E * 32;
    k_scatter<OUTC><<<(int)((tot + B - 1) / B), B, 0, stream>>>(hw, src, dst, coef, out, E);
  }
}
